// MambaLayer_81827716923671
// MI455X (gfx1250) — matrix operand order repaired and checked by reading, not compiled, not run on hardware
//
#include <hip/hip_runtime.h>
#include <hip/hip_bf16.h>

typedef __bf16 bf16_t;
typedef __attribute__((ext_vector_type(16))) __bf16 v16bf;
typedef __attribute__((ext_vector_type(8)))  float  v8f;
typedef int ai4 __attribute__((vector_size(4 * sizeof(int))));   // matches builtin's V4i

#define D_MODEL 1024
#define D_STATE 16
#define D_CONV  4
#define D_INNER 2048
#define BATCH   4
#define SEQ     2048
#define ROWS    (BATCH * SEQ)   // 8192 token rows

// ---- CDNA5 async global->LDS copy (ASYNCcnt-tracked), with sync fallback ----
#if defined(__has_builtin)
#  if __has_builtin(__builtin_amdgcn_global_load_async_to_lds_b128) && \
      __has_builtin(__builtin_amdgcn_s_wait_asynccnt)
#    define ATHENA_ASYNC_LDS 1
#  endif
#endif

__device__ __forceinline__ void cp16_g2l(uint4* lds, const uint4* g) {
#ifdef ATHENA_ASYNC_LDS
  __builtin_amdgcn_global_load_async_to_lds_b128(
      (__attribute__((address_space(1))) ai4*)g,     // global source
      (__attribute__((address_space(3))) ai4*)lds,   // LDS destination
      /*imm offset*/0, /*cpol*/0);
#else
  *lds = *g;
#endif
}

__device__ __forceinline__ void async_fence() {
#ifdef ATHENA_ASYNC_LDS
  __builtin_amdgcn_s_wait_asynccnt(0);
#endif
}

// ---------------------------------------------------------------------------
// fp32 -> bf16 pack
// ---------------------------------------------------------------------------
__global__ __launch_bounds__(256) void pack_bf16(const float* __restrict__ src,
                                                 bf16_t* __restrict__ dst, int n) {
  int i = blockIdx.x * 256 + threadIdx.x;
  if (i < n) dst[i] = (bf16_t)src[i];
}

// ---------------------------------------------------------------------------
// Tiled WMMA GEMM:  C[M,N] = A[M,K] * B[N,K]^T   (both operands K-major bf16)
// Block tile 128x256, BK=32, 256 threads = 8 waves (wave32).
// Wave grid 2(M) x 4(N); each wave owns a 64x64 macro-tile = 4x4 wmma tiles
// (16 ds_load_b128 : 16 v_wmma per k-step, 1:1).
// Double-buffered LDS; staging via GLOBAL_LOAD_ASYNC_TO_LDS_B128 overlapped
// with WMMA compute, synchronized with s_wait_asynccnt + barrier.
// EPI==0: plain fp32 store.  EPI==1: cols<2048 -> x_p fp32, cols>=2048 -> silu->bf16.
// ---------------------------------------------------------------------------
template <int EPI>
__global__ __launch_bounds__(256) void gemm_wmma_bf16(
    const bf16_t* __restrict__ A, const bf16_t* __restrict__ B,
    float* __restrict__ Cf, bf16_t* __restrict__ Cg,
    int M, int N, int K)
{
  constexpr int LDT = 40;                    // padded LDS row stride (bf16): 80B, 16B-mult
  __shared__ bf16_t sA[2][128 * LDT];        // 2 x 10 KB
  __shared__ bf16_t sB[2][256 * LDT];        // 2 x 20 KB

  const int tid  = threadIdx.x;
  const int lane = tid & 31;
  const int wave = tid >> 5;
  const int wm   = (wave & 1) * 64;          // wave M offset inside block tile
  const int wn   = (wave >> 1) * 64;         // wave N offset inside block tile
  const int bm   = blockIdx.y * 128;
  const int bn   = blockIdx.x * 256;
  const int half = lane >> 4;                // lane group (0: lanes 0-15, 1: 16-31)
  const int l16  = lane & 15;

  const v8f vzero = {0.f,0.f,0.f,0.f,0.f,0.f,0.f,0.f};
  v8f acc[4][4];
#pragma unroll
  for (int mi = 0; mi < 4; ++mi)
#pragma unroll
    for (int ni = 0; ni < 4; ++ni) acc[mi][ni] = vzero;

  const uint4* Agv = (const uint4*)A;        // 8 bf16 per uint4
  const uint4* Bgv = (const uint4*)B;
  const int Kv = K >> 3;

  // stage one 128x32 A tile + 256x32 B tile into LDS buffer `buf` (async DMA)
  auto stage = [&](int buf, int kt) {
    const int kv = kt >> 3;
#pragma unroll
    for (int i = 0; i < 2; ++i) {            // A: 512 x 16B chunks
      int idx = tid + i * 256;
      int row = idx >> 2, c = idx & 3;
      cp16_g2l((uint4*)(sA[buf] + row * LDT) + c,
               Agv + (size_t)(bm + row) * Kv + kv + c);
    }
#pragma unroll
    for (int i = 0; i < 4; ++i) {            // B: 1024 x 16B chunks
      int idx = tid + i * 256;
      int row = idx >> 2, c = idx & 3;
      cp16_g2l((uint4*)(sB[buf] + row * LDT) + c,
               Bgv + (size_t)(bn + row) * Kv + kv + c);
    }
  };

  union FragU { struct { uint4 lo; uint4 hi; } u; v16bf v; };

  const int nk = K / 32;
  stage(0, 0);
  async_fence();
  __syncthreads();

  for (int t = 0; t < nk; ++t) {
    const int cur = t & 1;
    if (t + 1 < nk) stage(cur ^ 1, (t + 1) * 32);   // async: lands during compute
    if (t + 2 < nk && wave == 0) {                  // L2 prefetch two tiles ahead
      __builtin_prefetch(&A[(size_t)(bm + lane * 4) * K + (t + 2) * 32], 0, 1);
      __builtin_prefetch(&B[(size_t)(bn + lane * 8) * K + (t + 2) * 32], 0, 1);
    }

    // ---- WMMA fragments per CDNA5 16-bit A(16x32)/B(32x16) lane layouts ----
    // A lane l: row M=l%16; elems 0..7 = K[8h..8h+7], elems 8..15 = K[16+8h..+7]
    FragU af[4];
#pragma unroll
    for (int mi = 0; mi < 4; ++mi) {
      const bf16_t* rp = sA[cur] + (wm + mi * 16 + l16) * LDT;
      af[mi].u.lo = *(const uint4*)(rp + 8 * half);
      af[mi].u.hi = *(const uint4*)(rp + 16 + 8 * half);
    }
    // B lane l: col N=l%16; elems 0..7 = K[8h..8h+7], elems 8..15 = K[16+8h..+7]
    FragU bfr[4];
#pragma unroll
    for (int ni = 0; ni < 4; ++ni) {
      const bf16_t* rp = sB[cur] + (wn + ni * 16 + l16) * LDT;
      bfr[ni].u.lo = *(const uint4*)(rp + 8 * half);
      bfr[ni].u.hi = *(const uint4*)(rp + 16 + 8 * half);
    }

#pragma unroll
    for (int mi = 0; mi < 4; ++mi)
#pragma unroll
      for (int ni = 0; ni < 4; ++ni)
        acc[mi][ni] = __builtin_amdgcn_wmma_f32_16x16x32_bf16(
            false, af[mi].v, false, bfr[ni].v, (short)0, acc[mi][ni], false, false);

    async_fence();        // next-buffer DMA landed
    __syncthreads();      // all waves done reading `cur`, LDS writes visible
  }

  // ---- epilogue: C/D layout -> lane l holds N=l%16, VGPR r -> M = 8*half + r ----
#pragma unroll
  for (int mi = 0; mi < 4; ++mi) {
#pragma unroll
    for (int ni = 0; ni < 4; ++ni) {
      int ng = bn + wn + ni * 16 + l16;
#pragma unroll
      for (int r = 0; r < 8; ++r) {
        int mg = bm + wm + mi * 16 + half * 8 + r;
        float v = acc[mi][ni][r];
        if (EPI == 0) {
          Cf[(size_t)mg * N + ng] = v;
        } else {
          if (ng < D_INNER) {
            Cf[(size_t)mg * D_INNER + ng] = v;                 // x_p branch (fp32)
          } else {
            float s = v / (1.0f + __expf(-v));                 // silu(res) -> bf16
            Cg[(size_t)mg * D_INNER + (ng - D_INNER)] = (bf16_t)s;
          }
        }
      }
    }
  }
}

// ---------------------------------------------------------------------------
// Causal depthwise conv1d (k=4) + bias + SiLU.  One thread per (b, d), 128-step
// s-chunk with a register shift window.
// ---------------------------------------------------------------------------
__global__ __launch_bounds__(256) void conv_silu(
    const float* __restrict__ xp, const float* __restrict__ w,
    const float* __restrict__ bias, float* __restrict__ xc)
{
  int d  = blockIdx.x * 256 + threadIdx.x;
  int s0 = blockIdx.y * 128;
  int b  = blockIdx.z;
  const float w0 = w[d * 4 + 0], w1 = w[d * 4 + 1], w2 = w[d * 4 + 2], w3 = w[d * 4 + 3];
  const float bb = bias[d];
  size_t base = ((size_t)b * SEQ) * D_INNER + d;
  float xm3 = (s0 >= 3) ? xp[base + (size_t)(s0 - 3) * D_INNER] : 0.f;
  float xm2 = (s0 >= 2) ? xp[base + (size_t)(s0 - 2) * D_INNER] : 0.f;
  float xm1 = (s0 >= 1) ? xp[base + (size_t)(s0 - 1) * D_INNER] : 0.f;
  for (int s = s0; s < s0 + 128; ++s) {
    float cur = xp[base + (size_t)s * D_INNER];
    float a = fmaf(w0, xm3, fmaf(w1, xm2, fmaf(w2, xm1, fmaf(w3, cur, bb))));
    xc[base + (size_t)s * D_INNER] = a / (1.0f + __expf(-a));
    xm3 = xm2; xm2 = xm1; xm1 = cur;
  }
}

// ---------------------------------------------------------------------------
// x_proj: BC[row, 0..31] = xc[row,:] . x_proj_w[j,:]   (K=2048, tiny GEMM)
// ---------------------------------------------------------------------------
__global__ __launch_bounds__(256) void xproj(
    const float* __restrict__ xc, const float* __restrict__ wp,
    float* __restrict__ bc)
{
  __shared__ float srow[D_INNER];
  __shared__ float part[8][32];
  int row = blockIdx.x;
  const float* xr = xc + (size_t)row * D_INNER;
  for (int i = threadIdx.x; i < D_INNER; i += 256) srow[i] = xr[i];
  __syncthreads();
  int j = threadIdx.x & 31;
  int p = threadIdx.x >> 5;
  const float* wr = wp + (size_t)j * D_INNER + p * 256;
  const float* sr = srow + p * 256;
  float acc = 0.f;
#pragma unroll 8
  for (int k = 0; k < 256; ++k) acc = fmaf(sr[k], wr[k], acc);
  part[p][j] = acc;
  __syncthreads();
  if (threadIdx.x < 32) {
    float t = 0.f;
#pragma unroll
    for (int q = 0; q < 8; ++q) t += part[q][threadIdx.x];
    bc[(size_t)row * 32 + threadIdx.x] = t;
  }
}

// ---------------------------------------------------------------------------
// Selective scan: thread per (b, d), h[16]/decay[16] in registers, B/C broadcast
// via LDS each step. Fuses gating y * silu(res) -> bf16, writes h_final.
// ---------------------------------------------------------------------------
__global__ __launch_bounds__(256) void scan_kernel(
    const float* __restrict__ xc, const float* __restrict__ bc,
    const float* __restrict__ A_log, const float* __restrict__ Dp,
    const float* __restrict__ state0, const bf16_t* __restrict__ resg,
    bf16_t* __restrict__ ybuf, float* __restrict__ hfinal)
{
  __shared__ float sbc[32];
  int d = blockIdx.x * 256 + threadIdx.x;
  int b = blockIdx.y;
  float h[D_STATE], dec[D_STATE];
#pragma unroll
  for (int n = 0; n < D_STATE; ++n) {
    dec[n] = __expf(-__expf(A_log[d * D_STATE + n]));
    h[n]   = state0[((size_t)b * D_INNER + d) * D_STATE + n];
  }
  const float Dv = Dp[d];
  const size_t rowbase = (size_t)b * SEQ;
  for (int s = 0; s < SEQ; ++s) {
    size_t row = rowbase + s;
    __syncthreads();
    if (threadIdx.x < 32) sbc[threadIdx.x] = bc[row * 32 + threadIdx.x];
    __syncthreads();
    float x = xc[row * D_INNER + d];
    float y = Dv * x;
#pragma unroll
    for (int n = 0; n < D_STATE; ++n) {
      h[n] = fmaf(h[n], dec[n], x * sbc[n]);        // h = h*decay + x*B
      y    = fmaf(h[n], sbc[16 + n], y);            // y += h*C
    }
    float g = (float)resg[row * D_INNER + d];       // silu(res), precomputed
    ybuf[row * D_INNER + d] = (bf16_t)(y * g);
  }
#pragma unroll
  for (int n = 0; n < D_STATE; ++n)
    hfinal[((size_t)b * D_INNER + d) * D_STATE + n] = h[n];
}

// ---------------------------------------------------------------------------
extern "C" void kernel_launch(void* const* d_in, const int* in_sizes, int n_in,
                              void* d_out, int out_size, void* d_ws, size_t ws_size,
                              hipStream_t stream)
{
  (void)in_sizes; (void)n_in; (void)out_size; (void)ws_size;
  const float* x          = (const float*)d_in[0];
  const float* state      = (const float*)d_in[1];
  const float* in_proj_w  = (const float*)d_in[2];
  const float* conv_w     = (const float*)d_in[3];
  const float* conv_b     = (const float*)d_in[4];
  const float* x_proj_w   = (const float*)d_in[5];
  const float* A_log      = (const float*)d_in[6];
  const float* Dvec       = (const float*)d_in[7];
  const float* out_proj_w = (const float*)d_in[8];

  char* ws = (char*)d_ws;
  size_t off = 0;
  auto alloc = [&](size_t bytes) -> void* {
    void* p = ws + off;
    off = (off + bytes + 255) & ~(size_t)255;
    return p;
  };
  bf16_t* xb   = (bf16_t*)alloc((size_t)ROWS * D_MODEL * 2);       // x in bf16
  bf16_t* w1b  = (bf16_t*)alloc((size_t)2 * D_INNER * D_MODEL * 2);// in_proj_w bf16
  bf16_t* w3b  = (bf16_t*)alloc((size_t)D_MODEL * D_INNER * 2);    // out_proj_w bf16
  float*  xp   = (float*)alloc((size_t)ROWS * D_INNER * 4);        // pre-conv branch
  bf16_t* resg = (bf16_t*)alloc((size_t)ROWS * D_INNER * 2);       // silu(res)
  float*  xcb  = (float*)alloc((size_t)ROWS * D_INNER * 4);        // post conv+silu
  float*  bcb  = (float*)alloc((size_t)ROWS * 32 * 4);             // B|C per row
  bf16_t* ybuf = (bf16_t*)alloc((size_t)ROWS * D_INNER * 2);       // gated y

  float* out  = (float*)d_out;
  float* hfin = out + (size_t)ROWS * D_MODEL;                      // second output

  // 1) pack fp32 -> bf16
  pack_bf16<<<(ROWS * D_MODEL + 255) / 256, 256, 0, stream>>>(x, xb, ROWS * D_MODEL);
  pack_bf16<<<(2 * D_INNER * D_MODEL + 255) / 256, 256, 0, stream>>>(in_proj_w, w1b, 2 * D_INNER * D_MODEL);
  pack_bf16<<<(D_MODEL * D_INNER + 255) / 256, 256, 0, stream>>>(out_proj_w, w3b, D_MODEL * D_INNER);

  // 2) in_proj GEMM (M=8192, N=4096, K=1024) with split/silu epilogue
  gemm_wmma_bf16<1><<<dim3((2 * D_INNER) / 256, ROWS / 128), 256, 0, stream>>>(
      xb, w1b, xp, resg, ROWS, 2 * D_INNER, D_MODEL);

  // 3) causal depthwise conv + SiLU
  conv_silu<<<dim3(D_INNER / 256, SEQ / 128, BATCH), 256, 0, stream>>>(xp, conv_w, conv_b, xcb);

  // 4) x_proj -> B,C
  xproj<<<ROWS, 256, 0, stream>>>(xcb, x_proj_w, bcb);

  // 5) selective scan + gating (+ h_final)
  scan_kernel<<<dim3(D_INNER / 256, BATCH), 256, 0, stream>>>(
      xcb, bcb, A_log, Dvec, state, resg, ybuf, hfin);

  // 6) out_proj GEMM (M=8192, N=1024, K=2048) -> fp32 output
  gemm_wmma_bf16<0><<<dim3(D_MODEL / 256, ROWS / 128), 256, 0, stream>>>(
      ybuf, w3b, out, nullptr, ROWS, D_MODEL, D_INNER);
}
